// CrossAttentionUnitDeltaWriterV3434_84791244358061
// MI455X (gfx1250) — matrix operand order repaired and checked by reading, not compiled, not run on hardware
//
#include <hip/hip_runtime.h>
#include <math.h>

typedef __attribute__((ext_vector_type(16))) _Float16 v16h;
typedef __attribute__((ext_vector_type(8)))  _Float16 v8h;
typedef __attribute__((ext_vector_type(8)))  float    v8f;

#define B_    4
#define M_    512
#define H_    4
#define K_    8
#define U_    64
#define SEM_  768
#define TRC_  512
#define HID_  256
#define QIN_  1027
#define QINP_ 1056
#define FIN_  1284
#define FINP_ 1312
#define XR_   65536     /* b*m*h*k rows */

#define OUT_DELTA 0
#define OUT_ENT   786432
#define OUT_MAX   787456
#define OUT_MASS  788480
#define OUT_Q     789504
#define OUT_CTX   1051648

// ---------------- WMMA helpers (CDNA5 16x16x32 f16 -> f32) ----------------

__device__ __forceinline__ v8f wmma16(v16h a, v16h b, v8f c) {
  return __builtin_amdgcn_wmma_f32_16x16x32_f16(false, a, false, b, (short)0, c, false, false);
}

// A fragment [16 x 32] from row-major f16: contiguous k = 16*(lane/16) + s, two aligned 16B chunks per lane.
__device__ __forceinline__ v16h load_a_rm(const _Float16* A, int lda, int k0, int lane) {
  const _Float16* p = A + (size_t)(lane & 15) * lda + k0 + ((lane >> 4) << 4);
  v8h lo = *(const v8h*)p;
  v8h hi = *(const v8h*)(p + 8);
  return __builtin_shufflevector(lo, hi, 0, 1, 2, 3, 4, 5, 6, 7, 8, 9, 10, 11, 12, 13, 14, 15);
}

// Contiguous 32B fragment load (pre-swizzled B tiles / key rows / valT rows).
__device__ __forceinline__ v16h ld16c(const _Float16* p) { return *(const v16h*)p; }

// Pre-swizzled weight tile: offset of tile (k0/32, n0/16), this lane's 16 halfs.
__device__ __forceinline__ v16h load_b_swz(const _Float16* W, int KT, int k0, int n0, int lane) {
  return ld16c(W + ((size_t)((n0 >> 4) * KT + (k0 >> 5)) * 32 + lane) * 16);
}

// element (k=i, n=j) -> swizzled offset (must match load_b_swz)
__device__ __forceinline__ size_t swz_off(int i, int j, int KT) {
  int nt = j >> 4, kt = i >> 5, kr = i & 31;
  int lane = ((kr >> 4) << 4) | (j & 15);
  return ((size_t)(nt * KT + kt) * 32 + lane) * 16 + (kr & 15);
}

__device__ __forceinline__ float geluf(float x) { return 0.5f * x * (1.0f + erff(x * 0.70710678118f)); }
__device__ __forceinline__ float ntn(float v) { return (v == v && fabsf(v) < 3.0e38f) ? v : 0.0f; }

// ---------------- LayerNorm row helpers (block = 128 thr, 16 rows x 8 thr) ----------------

__device__ void ln_rows_f32(const float* src, int rowpitch, int din, _Float16* dst, int dpitch, int do_ntn) {
  int row = threadIdx.x >> 3, sub = threadIdx.x & 7;
  const float* p = src + (size_t)row * rowpitch;
  _Float16* d = dst + (size_t)row * dpitch;
  float s = 0.f, ss = 0.f;
  for (int i = sub; i < din; i += 8) {
    float v = p[i]; if (do_ntn) v = ntn(v);
    d[i] = (_Float16)v; s += v; ss += v * v;
  }
  s += __shfl_xor(s, 1); s += __shfl_xor(s, 2); s += __shfl_xor(s, 4);
  ss += __shfl_xor(ss, 1); ss += __shfl_xor(ss, 2); ss += __shfl_xor(ss, 4);
  float mu = s / din, var = ss / din - mu * mu, rs = rsqrtf(fmaxf(var, 0.f) + 1e-5f);
  for (int i = sub; i < din; i += 8) { float v = (float)d[i]; d[i] = (_Float16)((v - mu) * rs); }
}

__device__ void ln_inplace_f16(_Float16* buf, int pitch, int din) {
  int row = threadIdx.x >> 3, sub = threadIdx.x & 7;
  _Float16* d = buf + (size_t)row * pitch;
  float s = 0.f, ss = 0.f;
  for (int i = sub; i < din; i += 8) { float v = (float)d[i]; s += v; ss += v * v; }
  s += __shfl_xor(s, 1); s += __shfl_xor(s, 2); s += __shfl_xor(s, 4);
  ss += __shfl_xor(ss, 1); ss += __shfl_xor(ss, 2); ss += __shfl_xor(ss, 4);
  float mu = s / din, var = ss / din - mu * mu, rs = rsqrtf(fmaxf(var, 0.f) + 1e-5f);
  for (int i = sub; i < din; i += 8) { float v = (float)d[i]; d[i] = (_Float16)((v - mu) * rs); }
}

// Generic 16-row WMMA GEMM, K-outer with A-fragment reuse across NT tiles per wave.
// dout = NT * 64. Wave handles n-tiles (wave + t*4), t in [0,NT).
template <int NT>
__device__ void gemm16t(const _Float16* A, int apitch, int Kdim, const _Float16* W,
                        const float* cvec, int act,
                        _Float16* qdst, int qpitch, int qcol0,
                        float* fdst, size_t fstride, int R0) {
  int wave = threadIdx.x >> 5, lane = threadIdx.x & 31;
  int KT = Kdim >> 5;
  v8f acc[NT];
#pragma unroll
  for (int t = 0; t < NT; ++t)
    for (int i = 0; i < 8; ++i) acc[t][i] = 0.f;
  for (int k0 = 0; k0 < Kdim; k0 += 32) {
    v16h a = load_a_rm(A, apitch, k0, lane);
#pragma unroll
    for (int t = 0; t < NT; ++t)
      acc[t] = wmma16(a, load_b_swz(W, KT, k0, (wave + t * 4) << 4, lane), acc[t]);
  }
#pragma unroll
  for (int t = 0; t < NT; ++t) {
    int n = ((wave + t * 4) << 4) + (lane & 15), mb = (lane >> 4) << 3;
    float c = cvec ? cvec[n] : 0.f;
    for (int i = 0; i < 8; ++i) {
      float v = acc[t][i] + c; if (act) v = geluf(v);
      if (qdst) qdst[(size_t)(mb + i) * qpitch + qcol0 + n] = (_Float16)v;
      if (fdst) fdst[((size_t)(R0 + mb + i)) * fstride + n] = v;
    }
  }
}

// ---------------- prep kernels ----------------

// fold LN affine into weights, emit swizzled f16 tiles + f32 column consts
__global__ void k_fold(const float* g, const float* bv, const float* W, const float* bias,
                       _Float16* Wf, float* cout, int din, int dinp, int dout) {
  int j = blockIdx.x, tid = threadIdx.x;
  int KT = dinp >> 5;
  __shared__ float red[256];
  float acc = 0.f;
  for (int i = tid; i < dinp; i += 256) {
    float w = (i < din) ? W[(size_t)i * dout + j] : 0.f;
    Wf[swz_off(i, j, KT)] = (_Float16)((i < din) ? g[i] * w : 0.f);
    if (i < din) acc += bv[i] * w;
  }
  red[tid] = acc; __syncthreads();
  for (int s = 128; s > 0; s >>= 1) { if (tid < s) red[tid] += red[tid + s]; __syncthreads(); }
  if (tid == 0) cout[j] = red[0] + bias[j];
}

// plain f32 -> swizzled f16 tiles (no LN fold)
__global__ void k_swz(const float* W, _Float16* Wf, int Kdim, int dout) {
  int j = blockIdx.x;
  int KT = Kdim >> 5;
  for (int i = threadIdx.x; i < Kdim; i += 256)
    Wf[swz_off(i, j, KT)] = (_Float16)W[(size_t)i * dout + j];
}

__global__ void k_conv(const float* src, _Float16* dst, int n) {
  for (int i = blockIdx.x * 256 + threadIdx.x; i < n; i += gridDim.x * 256) dst[i] = (_Float16)src[i];
}

// anchor = nan_to_num(obs).mean(2); point_evidence = sum_k ntn(raw)*max(w,1e-8); logw
__global__ void k_point(const float* obs, const float* raw, const float* tw,
                        float* anchor, float* pevid, float* logwv) {
  int bm = blockIdx.x, tid = threadIdx.x;
  __shared__ float wm[32];
  if (tid < 32) {
    float w = fmaxf(tw[(size_t)bm * 32 + tid], 1e-8f);
    wm[tid] = w; logwv[(size_t)bm * 32 + tid] = __logf(w);
  }
  const float* ob = obs + (size_t)bm * 4 * SEM_;
  for (int d = tid; d < SEM_; d += 256) {
    float s = 0.f; for (int j = 0; j < 4; ++j) s += ntn(ob[j * SEM_ + d]);
    anchor[(size_t)bm * SEM_ + d] = s * 0.25f;
  }
  __syncthreads();
  for (int h = 0; h < H_; ++h) {
    const float* rb = raw + ((size_t)bm * H_ + h) * K_ * SEM_;
    for (int d = tid; d < SEM_; d += 256) {
      float s = 0.f;
      for (int k = 0; k < K_; ++k) s += ntn(rb[(size_t)k * SEM_ + d]) * wm[h * K_ + k];
      pevid[((size_t)bm * H_ + h) * SEM_ + d] = s;
    }
  }
}

__global__ void k_logs(const float* assign, float* la, int n) {
  for (int i = blockIdx.x * 256 + threadIdx.x; i < n; i += gridDim.x * 256)
    la[i] = __logf(fmaxf(assign[i], 1e-8f));
}

// den, clipped sem-usage, assignment mass
__global__ void k_stats(const float* assign, const float* psu, float* denW,
                        float* semuclip, float* massclip, float* out) {
  int b = blockIdx.x, u = threadIdx.x;
  float ds = 0.f, hs[4] = {0.f, 0.f, 0.f, 0.f};
  for (int m = 0; m < M_; ++m) {
    float a = assign[((size_t)b * M_ + m) * U_ + u];
    ds += a;
    const float* pp = psu + ((size_t)b * M_ + m) * H_;
    for (int h = 0; h < H_; ++h) hs[h] += a * pp[h];
  }
  float den = fmaxf(ds, 1e-6f);
  denW[b * U_ + u] = den;
  for (int h = 0; h < H_; ++h) {
    int r = ((b * U_ + u) * H_ + h);
    semuclip[r] = fminf(fmaxf(hs[h] / den, 0.f), 1.f);
    float mass = fmaxf(hs[h], 0.f) / den;
    out[OUT_MASS + r] = mass;
    massclip[r] = fminf(fmaxf(mass, 0.f), 1.f);
  }
}

// pool over m: unit_trace / unit_anchor / unit_evidence (combined D = 512+768+768)
__global__ __launch_bounds__(256) void k_pool(const float* assign, const float* fth,
                                              const float* anchor, const float* pevid,
                                              const float* denW,
                                              float* ut, float* ua, float* ue) {
  int idx = blockIdx.x, bh = idx >> 3, dchunk = idx & 7;
  int b = bh >> 2, h = bh & 3;
  int d = dchunk * 256 + threadIdx.x;
  __shared__ float arow[64];
  float acc[64];
#pragma unroll
  for (int u = 0; u < 64; ++u) acc[u] = 0.f;
  for (int m = 0; m < M_; ++m) {
    if (threadIdx.x < 64) arow[threadIdx.x] = assign[((size_t)b * M_ + m) * U_ + threadIdx.x];
    __syncthreads();
    float v;
    if (d < 512)       v = fth[(((size_t)b * M_ + m) * H_ + h) * TRC_ + d];
    else if (d < 1280) v = anchor[((size_t)b * M_ + m) * SEM_ + (d - 512)];
    else               v = pevid[(((size_t)b * M_ + m) * H_ + h) * SEM_ + (d - 1280)];
#pragma unroll
    for (int u = 0; u < 64; ++u) acc[u] += arow[u] * v;
    __syncthreads();
  }
  for (int u = 0; u < 64; ++u) {
    int r = ((b * U_ + u) * H_ + h);
    float o = acc[u] / denW[b * U_ + u];
    if (d < 512)       ut[(size_t)r * TRC_ + d] = o;
    else if (d < 1280) ua[(size_t)r * SEM_ + (d - 512)] = o;
    else               ue[(size_t)r * SEM_ + (d - 1280)] = o;
  }
}

// ---------------- key/value projection (dominant WMMA GEMM) ----------------
// keyA layout: [b][h][n=m*8+k][256]   (score B^T fragment = contiguous 32B)
// valT layout: [b][h][c=256][n=4096]  (context B fragment = contiguous 32B)

__global__ __launch_bounds__(128) void k_proj(const float* Xraw,
    const _Float16* WK, const float* cK, const _Float16* WV, const float* cV,
    _Float16* keyA, _Float16* valT) {
  __shared__ alignas(32) _Float16 Xs[16 * SEM_];
  int R0 = blockIdx.x * 16;
  // prefetch next block's 16x768 f32 tile into cache (global_prefetch_b8 path)
  if (blockIdx.x + 1 < gridDim.x)
    __builtin_prefetch(Xraw + (size_t)(R0 + 16) * SEM_ + (size_t)threadIdx.x * 96, 0, 1);
  ln_rows_f32(Xraw + (size_t)R0 * SEM_, SEM_, SEM_, Xs, SEM_, 1);
  __syncthreads();
  int wave = threadIdx.x >> 5, lane = threadIdx.x & 31;
  // K-outer: one A fragment feeds all 8 (kv, n-tile) jobs of this wave
  v8f acc[8];
#pragma unroll
  for (int jj = 0; jj < 8; ++jj)
    for (int i = 0; i < 8; ++i) acc[jj][i] = 0.f;
  for (int k0 = 0; k0 < SEM_; k0 += 32) {
    v16h a = load_a_rm(Xs, SEM_, k0, lane);
#pragma unroll
    for (int jj = 0; jj < 8; ++jj) {
      int j = wave + jj * 4;
      const _Float16* W = (j >> 4) ? WV : WK;
      acc[jj] = wmma16(a, load_b_swz(W, SEM_ >> 5, k0, (j & 15) << 4, lane), acc[jj]);
    }
  }
#pragma unroll
  for (int jj = 0; jj < 8; ++jj) {
    int j = wave + jj * 4, kv = j >> 4, nt = j & 15;
    const float* cv = kv ? cV : cK;
    int n = (nt << 4) + (lane & 15), mb = (lane >> 4) << 3;
    for (int i = 0; i < 8; ++i) {
      float v = acc[jj][i] + cv[n];
      int r = R0 + mb + i;                      // flat (b,m,h,k)
      int bb = r >> 14, mm = (r >> 5) & 511, hh = (r >> 3) & 3, kk = r & 7;
      size_t na = (size_t)mm * 8 + kk;
      size_t bh = (size_t)(bb * 4 + hh);
      if (kv) valT[(bh * 256 + n) * 4096 + na] = (_Float16)v;
      else    keyA[(bh * 4096 + na) * 256 + n] = (_Float16)v;
    }
  }
}

// ---------------- per-unit projections + query MLP ----------------

__global__ __launch_bounds__(128) void k_unit(
    const float* ut, const float* ua, const float* ue, const float* um,
    const float* uconf, const float* uusage, const float* semu,
    const _Float16* WT, const float* cT,
    const _Float16* WS, const float* cS,
    const _Float16* WO, const float* cO,
    const _Float16* WQ1, const float* cQ1,
    const _Float16* WQ2, const float* qb2,
    float* trace_h, float* anchor_h, float* old_h,
    float* outq, _Float16* qf16) {
  __shared__ alignas(32) _Float16 Qin[16 * QINP_];
  __shared__ alignas(32) _Float16 Ab[16 * SEM_];
  int R0 = blockIdx.x * 16;

  ln_rows_f32(ut + (size_t)R0 * TRC_, TRC_, TRC_, Ab, SEM_, 0);
  __syncthreads();
  gemm16t<4>(Ab, SEM_, TRC_, WT, cT, 0, Qin, QINP_, 0, trace_h, HID_, R0);
  __syncthreads();

  ln_rows_f32(ua + (size_t)R0 * SEM_, SEM_, SEM_, Ab, SEM_, 0);
  __syncthreads();
  gemm16t<4>(Ab, SEM_, SEM_, WS, cS, 0, Qin, QINP_, HID_, anchor_h, HID_, R0);
  __syncthreads();

  ln_rows_f32(ue + (size_t)R0 * SEM_, SEM_, SEM_, Ab, SEM_, 0);
  __syncthreads();
  gemm16t<4>(Ab, SEM_, SEM_, WS, cS, 0, Qin, QINP_, 2 * HID_, nullptr, 0, R0);
  __syncthreads();

  ln_rows_f32(um + (size_t)R0 * SEM_, SEM_, SEM_, Ab, SEM_, 0);
  __syncthreads();
  gemm16t<4>(Ab, SEM_, SEM_, WO, cO, 0, Qin, QINP_, 3 * HID_, old_h, HID_, R0);

  if (threadIdx.x < 16) {
    int r = R0 + threadIdx.x, b = r >> 8, u = (r >> 2) & 63, h = r & 3;
    _Float16* q = Qin + (size_t)threadIdx.x * QINP_;
    q[1024] = (_Float16)uconf[b * U_ + u];
    q[1025] = (_Float16)fminf(fmaxf(uusage[(b * U_ + u) * H_ + h], 0.f), 1.f);
    q[1026] = (_Float16)semu[r];
    for (int c = QIN_; c < QINP_; ++c) q[c] = (_Float16)0.f;
  }
  __syncthreads();
  ln_inplace_f16(Qin, QINP_, QIN_);
  __syncthreads();
  gemm16t<4>(Qin, QINP_, QINP_, WQ1, cQ1, 1, Ab, HID_, 0, nullptr, 0, R0);
  __syncthreads();
  gemm16t<4>(Ab, HID_, HID_, WQ2, qb2, 0, qf16 + (size_t)R0 * HID_, HID_, 0, outq, HID_, R0);
}

// ---------------- flash attention over m*k (N = 4096) ----------------

__global__ __launch_bounds__(256) void k_attn(
    const _Float16* keyA, const _Float16* valT, const _Float16* qf16,
    const float* logass, const float* logw, float* out) {
  __shared__ alignas(32) _Float16 P[4][16 * 32];
  int b = blockIdx.x >> 2, h = blockIdx.x & 3;
  int wave = threadIdx.x >> 5, lane = threadIdx.x & 31;
  int ut = wave >> 1, chalf = wave & 1, u0 = ut * 16;
  int mlo = lane & 15, hi = lane >> 4, b8 = hi << 3;

  const _Float16* qbase = qf16 + ((size_t)b * 65536 + (size_t)h * 256);
  const _Float16* kb = keyA + (size_t)(b * 4 + h) * 4096 * 256;
  const _Float16* vbT = valT + (size_t)(b * 4 + h) * 256 * 4096;

  v16h qa[8];
#pragma unroll
  for (int ks = 0; ks < 8; ++ks)
    qa[ks] = load_a_rm(qbase + (size_t)u0 * 1024, 1024, ks * 32, lane);

  float mrun[8], lrun[8], trun[8];
  for (int i = 0; i < 8; ++i) { mrun[i] = -3.0e38f; lrun[i] = 0.f; trun[i] = 0.f; }
  v8f acc[8];
  for (int c = 0; c < 8; ++c) for (int i = 0; i < 8; ++i) acc[c][i] = 0.f;
  _Float16* Pw = &P[ut][0];

  for (int ch = 0; ch < 128; ++ch) {
    int nb = ch * 32;
    float lg[2][8];
#pragma unroll
    for (int t = 0; t < 2; ++t) {
      int n0 = nb + t * 16;
      v8f sc; for (int i = 0; i < 8; ++i) sc[i] = 0.f;
      int n = n0 + mlo, mm = n >> 3, kk = n & 7;
      const _Float16* krow = kb + (size_t)n * 256 + (hi << 4);
#pragma unroll
      for (int ks = 0; ks < 8; ++ks)
        sc = wmma16(qa[ks], ld16c(krow + ks * 32), sc);
      float lw = logw[(size_t)(b * M_ + mm) * 32 + h * 8 + kk];
      const float* lap = logass + (size_t)(b * M_ + mm) * U_ + u0 + b8;
#pragma unroll
      for (int i = 0; i < 8; ++i) lg[t][i] = sc[i] * 0.0625f + lw + lap[i];
    }
    // online softmax update (rows replicated across 16-lane group)
    float rm[8], scale8[8], ps[8], ts[8];
#pragma unroll
    for (int i = 0; i < 8; ++i) rm[i] = fmaxf(lg[0][i], lg[1][i]);
#pragma unroll
    for (int i = 0; i < 8; ++i) {
      rm[i] = fmaxf(rm[i], __shfl_xor(rm[i], 1));
      rm[i] = fmaxf(rm[i], __shfl_xor(rm[i], 2));
      rm[i] = fmaxf(rm[i], __shfl_xor(rm[i], 4));
      rm[i] = fmaxf(rm[i], __shfl_xor(rm[i], 8));
    }
#pragma unroll
    for (int i = 0; i < 8; ++i) {
      float mn = fmaxf(mrun[i], rm[i]);
      scale8[i] = __expf(mrun[i] - mn);
      float e0 = __expf(lg[0][i] - mn), e1 = __expf(lg[1][i] - mn);
      if (chalf == 0) {
        Pw[(size_t)(b8 + i) * 32 + mlo] = (_Float16)e0;
        Pw[(size_t)(b8 + i) * 32 + 16 + mlo] = (_Float16)e1;
      }
      ps[i] = e0 + e1;
      ts[i] = e0 * lg[0][i] + e1 * lg[1][i];
      mrun[i] = mn;
    }
#pragma unroll
    for (int i = 0; i < 8; ++i) {
      ps[i] += __shfl_xor(ps[i], 1); ps[i] += __shfl_xor(ps[i], 2);
      ps[i] += __shfl_xor(ps[i], 4); ps[i] += __shfl_xor(ps[i], 8);
      ts[i] += __shfl_xor(ts[i], 1); ts[i] += __shfl_xor(ts[i], 2);
      ts[i] += __shfl_xor(ts[i], 4); ts[i] += __shfl_xor(ts[i], 8);
      lrun[i] = lrun[i] * scale8[i] + ps[i];
      trun[i] = trun[i] * scale8[i] + ts[i];
    }
    __syncthreads();
    v16h pa = load_a_rm(Pw, 32, 0, lane);
#pragma unroll
    for (int cc = 0; cc < 8; ++cc) {
      int c = ((chalf * 8 + cc) << 4) + mlo;
      v16h bf = ld16c(vbT + (size_t)c * 4096 + nb + (hi << 4));
      v8f t = acc[cc];
#pragma unroll
      for (int i = 0; i < 8; ++i) t[i] *= scale8[i];
      acc[cc] = wmma16(pa, bf, t);
    }
    __syncthreads();
  }
  // finalize: entropy = m + log l - T/l ; max p = 1/l ; context /= l
  if (chalf == 0 && mlo == 0) {
    for (int i = 0; i < 8; ++i) {
      int r = ((b * U_ + u0 + b8 + i) * H_ + h);
      float l = fmaxf(lrun[i], 1e-30f);
      out[OUT_ENT + r] = mrun[i] + __logf(l) - trun[i] / l;
      out[OUT_MAX + r] = 1.0f / l;
    }
  }
  for (int cc = 0; cc < 8; ++cc) {
    int ct = chalf * 8 + cc;
    for (int i = 0; i < 8; ++i) {
      int r = ((b * U_ + u0 + b8 + i) * H_ + h);
      out[OUT_CTX + (size_t)r * HID_ + (ct << 4) + mlo] = acc[cc][i] / fmaxf(lrun[i], 1e-30f);
    }
  }
}

// ---------------- fuse MLP + direction/magnitude ----------------

__global__ __launch_bounds__(128) void k_fuse(
    const _Float16* qf16, const float* outbuf,
    const float* trace_h, const float* anchor_h, const float* old_h,
    const float* uusage, const float* massclip,
    const _Float16* WF1, const float* cF1,
    const _Float16* WF2, const float* fb2,
    const _Float16* WD, const float* db,
    const _Float16* WMw, const float* mbias,
    float* out) {
  __shared__ alignas(32) _Float16 Fin[16 * FINP_];
  __shared__ alignas(32) _Float16 Hb[16 * 512];
  __shared__ float dss[16];
  __shared__ float mag[16];
  int R0 = blockIdx.x * 16;
  int row = threadIdx.x >> 3, sub = threadIdx.x & 7;
  {
    int r = R0 + row, b = r >> 8, u = (r >> 2) & 63, h = r & 3;
    _Float16* d = Fin + (size_t)row * FINP_;
    const float* ctx = outbuf + OUT_CTX + (size_t)r * HID_;
    for (int i = sub; i < FINP_; i += 8) {
      float v;
      if (i < 256)       v = (float)qf16[(size_t)r * HID_ + i];
      else if (i < 512)  v = ctx[i - 256];
      else if (i < 768)  v = trace_h[(size_t)r * HID_ + i - 512];
      else if (i < 1024) v = anchor_h[(size_t)r * HID_ + i - 768];
      else if (i < 1280) v = old_h[(size_t)r * HID_ + i - 1024];
      else if (i == 1280) v = outbuf[OUT_ENT + r];
      else if (i == 1281) v = outbuf[OUT_MAX + r];
      else if (i == 1282) v = fminf(fmaxf(uusage[(b * U_ + u) * H_ + h], 0.f), 1.f);
      else if (i == 1283) v = massclip[r];
      else v = 0.f;
      d[i] = (_Float16)v;
    }
  }
  __syncthreads();
  ln_inplace_f16(Fin, FINP_, FIN_);
  if (threadIdx.x < 16) dss[threadIdx.x] = 0.f;
  __syncthreads();
  gemm16t<8>(Fin, FINP_, FINP_, WF1, cF1, 1, Hb, 512, 0, nullptr, 0, R0);
  __syncthreads();
  // layer 2, in-place on Hb (read-all then write), A fragments shared across 8 tiles
  {
    int wave = threadIdx.x >> 5, lane = threadIdx.x & 31;
    v8f acc2[8];
#pragma unroll
    for (int t = 0; t < 8; ++t)
      for (int i = 0; i < 8; ++i) acc2[t][i] = 0.f;
    for (int k0 = 0; k0 < 512; k0 += 32) {
      v16h a = load_a_rm(Hb, 512, k0, lane);
#pragma unroll
      for (int t = 0; t < 8; ++t)
        acc2[t] = wmma16(a, load_b_swz(WF2, 16, k0, (wave + t * 4) << 4, lane), acc2[t]);
    }
    __syncthreads();
    for (int t = 0; t < 8; ++t) {
      int n = ((wave + t * 4) << 4) + (lane & 15), mb2 = (lane >> 4) << 3;
      for (int i = 0; i < 8; ++i)
        Hb[(size_t)(mb2 + i) * 512 + n] = (_Float16)geluf(acc2[t][i] + fb2[n]);
    }
  }
  __syncthreads();
  // direction (768 cols = 12 tiles/wave) into Fin region + sum of squares
  {
    int wave = threadIdx.x >> 5, lane = threadIdx.x & 31;
    v8f accd[12];
#pragma unroll
    for (int t = 0; t < 12; ++t)
      for (int i = 0; i < 8; ++i) accd[t][i] = 0.f;
    for (int k0 = 0; k0 < 512; k0 += 32) {
      v16h a = load_a_rm(Hb, 512, k0, lane);
#pragma unroll
      for (int t = 0; t < 12; ++t)
        accd[t] = wmma16(a, load_b_swz(WD, 16, k0, (wave + t * 4) << 4, lane), accd[t]);
    }
#pragma unroll
    for (int t = 0; t < 12; ++t) {
      int n = ((wave + t * 4) << 4) + (lane & 15), mb2 = (lane >> 4) << 3;
      for (int i = 0; i < 8; ++i) {
        float v = accd[t][i] + db[n];
        Fin[(size_t)(mb2 + i) * FINP_ + n] = (_Float16)v;
        atomicAdd(&dss[mb2 + i], v * v);
      }
    }
  }
  // magnitude
  {
    float s = 0.f;
    const _Float16* hr = Hb + (size_t)row * 512;
    for (int i = sub; i < 512; i += 8) s += (float)hr[i] * (float)WMw[i];
    s += __shfl_xor(s, 1); s += __shfl_xor(s, 2); s += __shfl_xor(s, 4);
    if (sub == 0) mag[row] = 2.5f / (1.f + __expf(-(s + mbias[0])));
  }
  __syncthreads();
  for (int i = threadIdx.x; i < 16 * SEM_; i += 128) {
    int rr = i / SEM_, c = i - rr * SEM_;
    float scale = mag[rr] / fmaxf(sqrtf(dss[rr]), 1e-12f);
    out[OUT_DELTA + (size_t)(R0 + rr) * SEM_ + c] = (float)Fin[(size_t)rr * FINP_ + c] * scale;
  }
}

// ---------------- host ----------------

extern "C" void kernel_launch(void* const* d_in, const int* in_sizes, int n_in,
                              void* d_out, int out_size, void* d_ws, size_t ws_size,
                              hipStream_t stream) {
  (void)in_sizes; (void)n_in; (void)out_size; (void)ws_size;
  auto F = [&](int i) { return (const float*)d_in[i]; };
  float* out = (float*)d_out;

  char* w = (char*)d_ws;
  auto allocB = [&](size_t bytes) -> void* {
    void* p = (void*)w; w += (bytes + 255) & ~(size_t)255; return p;
  };
  // f16 buffers (weights swizzled into B-fragment tile order)
  _Float16* WT  = (_Float16*)allocB((size_t)512 * 256 * 2);
  _Float16* WKk = (_Float16*)allocB((size_t)768 * 256 * 2);
  _Float16* WVv = (_Float16*)allocB((size_t)768 * 256 * 2);
  _Float16* WSs = (_Float16*)allocB((size_t)768 * 256 * 2);
  _Float16* WOo = (_Float16*)allocB((size_t)768 * 256 * 2);
  _Float16* WQ1 = (_Float16*)allocB((size_t)QINP_ * 256 * 2);
  _Float16* WQ2 = (_Float16*)allocB((size_t)256 * 256 * 2);
  _Float16* WF1 = (_Float16*)allocB((size_t)FINP_ * 512 * 2);
  _Float16* WF2 = (_Float16*)allocB((size_t)512 * 512 * 2);
  _Float16* WD  = (_Float16*)allocB((size_t)512 * 768 * 2);
  _Float16* WMw = (_Float16*)allocB((size_t)512 * 2);
  _Float16* keyA = (_Float16*)allocB((size_t)XR_ * 256 * 2);
  _Float16* valT = (_Float16*)allocB((size_t)XR_ * 256 * 2);
  _Float16* qf16 = (_Float16*)allocB((size_t)1024 * 256 * 2);
  // f32 buffers
  float* cT  = (float*)allocB(256 * 4);
  float* cK  = (float*)allocB(256 * 4);
  float* cV  = (float*)allocB(256 * 4);
  float* cS  = (float*)allocB(256 * 4);
  float* cO  = (float*)allocB(256 * 4);
  float* cQ1 = (float*)allocB(256 * 4);
  float* cF1 = (float*)allocB(512 * 4);
  float* anchor  = (float*)allocB((size_t)B_ * M_ * SEM_ * 4);
  float* pevid   = (float*)allocB((size_t)B_ * M_ * H_ * SEM_ * 4);
  float* denW    = (float*)allocB(256 * 4);
  float* semuclip = (float*)allocB(1024 * 4);
  float* massclip = (float*)allocB(1024 * 4);
  float* logass  = (float*)allocB((size_t)B_ * M_ * U_ * 4);
  float* logw    = (float*)allocB((size_t)B_ * M_ * 32 * 4);
  float* ut = (float*)allocB((size_t)1024 * TRC_ * 4);
  float* ua = (float*)allocB((size_t)1024 * SEM_ * 4);
  float* ue = (float*)allocB((size_t)1024 * SEM_ * 4);
  float* trace_h  = (float*)allocB((size_t)1024 * 256 * 4);
  float* anchor_h = (float*)allocB((size_t)1024 * 256 * 4);
  float* old_h    = (float*)allocB((size_t)1024 * 256 * 4);

  // 1) fold LN+Linear weights into swizzled f16 tiles
  k_fold<<<256, 256, 0, stream>>>(F(9),  F(10), F(11), F(12), WT,  cT,  512,  512,  256);
  k_fold<<<256, 256, 0, stream>>>(F(13), F(14), F(15), F(16), WKk, cK,  768,  768,  256);
  k_fold<<<256, 256, 0, stream>>>(F(17), F(18), F(19), F(20), WVv, cV,  768,  768,  256);
  k_fold<<<256, 256, 0, stream>>>(F(21), F(22), F(23), F(24), WSs, cS,  768,  768,  256);
  k_fold<<<256, 256, 0, stream>>>(F(25), F(26), F(27), F(28), WOo, cO,  768,  768,  256);
  k_fold<<<256, 256, 0, stream>>>(F(29), F(30), F(31), F(32), WQ1, cQ1, QIN_, QINP_, 256);
  k_fold<<<512, 256, 0, stream>>>(F(35), F(36), F(37), F(38), WF1, cF1, FIN_, FINP_, 512);
  k_swz<<<256, 256, 0, stream>>>(F(33), WQ2, 256, 256);
  k_swz<<<512, 256, 0, stream>>>(F(39), WF2, 512, 512);
  k_swz<<<768, 256, 0, stream>>>(F(41), WD,  512, 768);
  k_conv<<<1, 256, 0, stream>>>(F(43), WMw, 512);

  // 2) point-level prep, logs, per-unit stats, pooling
  k_point<<<B_ * M_, 256, 0, stream>>>(F(4), F(2), F(3), anchor, pevid, logw);
  k_logs<<<512, 256, 0, stream>>>(F(0), logass, B_ * M_ * U_);
  k_stats<<<B_, U_, 0, stream>>>(F(0), F(8), denW, semuclip, massclip, out);
  k_pool<<<B_ * H_ * 8, 256, 0, stream>>>(F(0), F(1), anchor, pevid, denW, ut, ua, ue);

  // 3) key/value projection (dominant WMMA GEMM)
  k_proj<<<XR_ / 16, 128, 0, stream>>>(F(2), WKk, cK, WVv, cV, keyA, valT);

  // 4) per-unit projections + query MLP
  k_unit<<<64, 128, 0, stream>>>(ut, ua, ue, F(5), F(6), F(7), semuclip,
                                 WT, cT, WSs, cS, WOo, cO, WQ1, cQ1, WQ2, F(34),
                                 trace_h, anchor_h, old_h, out + OUT_Q, qf16);

  // 5) flash attention over m*k
  k_attn<<<B_ * H_, 256, 0, stream>>>(keyA, valT, qf16, logass, logw, out);

  // 6) fuse MLP, direction, magnitude, delta
  k_fuse<<<64, 128, 0, stream>>>(qf16, (const float*)d_out, trace_h, anchor_h, old_h,
                                 F(7), massclip, WF1, cF1, WF2, F(40), WD, F(42),
                                 WMw, F(44), out);
}
